// IzhikevichSNNLayer_10153302687890
// MI455X (gfx1250) — matrix operand order repaired and checked by reading, not compiled, not run on hardware
//
#include <hip/hip_runtime.h>
#include <hip/hip_bf16.h>
#include <math.h>

// ---------------------------------------------------------------------------
// Izhikevich SNN layer, fused for gfx1250 (MI455X).
//   raw = x @ W.T + b  -> LayerNorm(2048) -> tanh * |mag| -> 8-step Izhikevich
// GEMM: M=4096, N=2048, K=1024 via v_wmma_f32_16x16x32_bf16.
// Block = 256 threads (8 waves) owns 16 rows x all 2048 cols (LN block-local).
// B fragments pipelined in groups of 4 (2-deep), pinned with sched_barrier so
// 8 loads stay in flight under each WMMA group (wait loadcnt<=8, not 0).
// ---------------------------------------------------------------------------

#define B_ROWS   4096
#define IN_F     1024
#define OUT_F    2048
#define BM       16          // rows per block (one WMMA M-tile)
#define NWAVES   8           // waves per block (256 threads, wave32)
#define NT_WAVE  16          // 16-col N-tiles per wave (16*16*8 = 2048)
#define GRP      4           // N-tiles per pipeline group

typedef __attribute__((ext_vector_type(16))) __bf16 v16bf;
typedef __attribute__((ext_vector_type(8)))  float  v8f;

union Frag16 {
    v16bf v;
    uint4 q[2];
};

__device__ __forceinline__ unsigned bf16rne(float f) {
    union { float f; unsigned u; } c;
    c.f = f;
    unsigned u = c.u;
    u += 0x7FFFu + ((u >> 16) & 1u);   // round-to-nearest-even
    return u >> 16;
}

__device__ __forceinline__ unsigned pack2bf(float a, float b) {
    return bf16rne(a) | (bf16rne(b) << 16);
}

__device__ __forceinline__ unsigned hashu(unsigned x) {
    x ^= x >> 16; x *= 0x7FEB352Du;
    x ^= x >> 15; x *= 0x846CA68Bu;
    x ^= x >> 16;
    return x;
}

// ---------------------------------------------------------------------------
// Kernel 0: W fp32 [2048,1024] -> bf16 in workspace (4 MB). Run once per call.
// ---------------------------------------------------------------------------
__global__ void wconv_kernel(const float* __restrict__ W,
                             unsigned short* __restrict__ Wb) {
    int i = (blockIdx.x * blockDim.x + threadIdx.x) * 8;   // 8 elems / thread
    const float4* s = (const float4*)(W + i);
    float4 f0 = s[0];
    float4 f1 = s[1];
    uint4 p = make_uint4(pack2bf(f0.x, f0.y), pack2bf(f0.z, f0.w),
                         pack2bf(f1.x, f1.y), pack2bf(f1.z, f1.w));
    *(uint4*)(Wb + i) = p;
}

// ---------------------------------------------------------------------------
// Kernel 1: fused GEMM + LayerNorm + tanh + Izhikevich.
// ---------------------------------------------------------------------------
__global__ __launch_bounds__(NWAVES * 32, 1)
void snn_fused_kernel(const float* __restrict__ x,
                      const unsigned short* __restrict__ Wb,   // bf16 bits
                      const float* __restrict__ b_lin,
                      const float* __restrict__ gamma,
                      const float* __restrict__ beta,
                      const float* __restrict__ cmag,
                      float* __restrict__ out) {
    // 32 KB static LDS: x tile as bf16, later reused for LN reduction.
    __shared__ __align__(16) unsigned short xs[BM * IN_F];

    const int tid   = threadIdx.x;
    const int wave  = tid >> 5;
    const int lane  = tid & 31;
    const int lm    = lane & 15;      // column index within 16x16 tile
    const int hh    = lane >> 4;      // half-wave id
    const int row0  = blockIdx.x * BM;
    const int wn0   = wave * (NT_WAVE * 16);   // first column of this wave

    // ---- stage x tile (16 x 1024 fp32 -> bf16) into LDS -------------------
    for (int i = tid; i < (BM * IN_F) / 8; i += NWAVES * 32) {
        int e = i * 8;
        int r = e >> 10;         // / IN_F
        int c = e & (IN_F - 1);
        const float4* src = (const float4*)(x + (size_t)(row0 + r) * IN_F + c);
        float4 f0 = src[0];
        float4 f1 = src[1];
        uint4 p = make_uint4(pack2bf(f0.x, f0.y), pack2bf(f0.z, f0.w),
                             pack2bf(f1.x, f1.y), pack2bf(f1.z, f1.w));
        *(uint4*)(xs + e) = p;
    }
    __syncthreads();

    // ---- GEMM: 16 N-tiles per wave, K-loop of 32, grouped B pipeline ------
    v8f acc[NT_WAVE] = {};

    // per-wave base pointers (tt / k offsets become instruction immediates)
    const unsigned short* bbase = Wb + (size_t)(wn0 + lm) * IN_F + hh * 16;
    const unsigned short* abase = xs + lm * IN_F + hh * 16;

    for (int kc = 0; kc < IN_F / 32; ++kc) {
        const int k0 = kc * 32;
        // A fragment (contiguous k order: lane<16 holds K 0..15)
        Frag16 a0;
        {
            const unsigned short* ap = abase + k0;
            a0.q[0] = *(const uint4*)(ap);
            a0.q[1] = *(const uint4*)(ap + 8);
        }
        // B fragment pipeline: 2 groups of 4 tiles in flight
        Frag16 bq[2][GRP];
#pragma unroll
        for (int j = 0; j < GRP; ++j) {
            const unsigned short* bp = bbase + (size_t)j * 16 * IN_F + k0;
            bq[0][j].q[0] = *(const uint4*)(bp);
            bq[0][j].q[1] = *(const uint4*)(bp + 8);
        }
#pragma unroll
        for (int g = 0; g < NT_WAVE / GRP; ++g) {
            if (g + 1 < NT_WAVE / GRP) {
#pragma unroll
                for (int j = 0; j < GRP; ++j) {
                    const unsigned short* bp =
                        bbase + (size_t)((g + 1) * GRP + j) * 16 * IN_F + k0;
                    bq[(g + 1) & 1][j].q[0] = *(const uint4*)(bp);
                    bq[(g + 1) & 1][j].q[1] = *(const uint4*)(bp + 8);
                }
                // Pin issue order: group g+1 loads stay BEFORE group g WMMAs,
                // so 8 loads remain in flight under the WMMA group.
                __builtin_amdgcn_sched_barrier(0);
            }
#pragma unroll
            for (int j = 0; j < GRP; ++j) {
                acc[g * GRP + j] = __builtin_amdgcn_wmma_f32_16x16x32_bf16(
                    false, a0.v, false, bq[g & 1][j].v, (short)0,
                    acc[g * GRP + j], false, false);
            }
        }
    }

    // ---- add bias, per-lane partial row sums ------------------------------
    float psum[8], psq[8];
#pragma unroll
    for (int j = 0; j < 8; ++j) { psum[j] = 0.f; psq[j] = 0.f; }

#pragma unroll
    for (int tt = 0; tt < NT_WAVE; ++tt) {
        const int n = wn0 + tt * 16 + lm;
        const float bl = b_lin[n];
#pragma unroll
        for (int r = 0; r < 8; ++r) {
            float v = acc[tt][r] + bl;
            acc[tt][r] = v;
            psum[r] += v;
            psq[r]  += v * v;
        }
    }
    // reduce across the 16 lanes that share rows (lanes with same hh)
#pragma unroll
    for (int off = 1; off < 16; off <<= 1) {
#pragma unroll
        for (int j = 0; j < 8; ++j) {
            psum[j] += __shfl_xor(psum[j], off, 16);
            psq[j]  += __shfl_xor(psq[j],  off, 16);
        }
    }

    // ---- block-level LN reduction in LDS (reuse xs after barrier) ---------
    __syncthreads();                       // all waves done reading xs
    float* red = (float*)xs;               // [0..15] sum, [16..31] sumsq,
                                           // [32..47] mu,  [48..63] rstd
    if (tid < 32) red[tid] = 0.f;
    __syncthreads();
    if (lm == 0) {
#pragma unroll
        for (int j = 0; j < 8; ++j) {
            int m = hh * 8 + j;
            atomicAdd(&red[m],      psum[j]);
            atomicAdd(&red[16 + m], psq[j]);
        }
    }
    __syncthreads();
    if (tid < 16) {
        float mu  = red[tid] * (1.0f / OUT_F);
        float var = red[16 + tid] * (1.0f / OUT_F) - mu * mu;
        red[32 + tid] = mu;
        red[48 + tid] = rsqrtf(var + 1e-5f);
    }
    __syncthreads();

    // ---- epilogue: LN -> tanh -> 8-step Izhikevich ------------------------
    const float mag = fabsf(cmag[0]);
#pragma unroll
    for (int tt = 0; tt < NT_WAVE; ++tt) {
        const int n = wn0 + tt * 16 + lm;
        const float g  = gamma[n];
        const float bb = beta[n];
#pragma unroll
        for (int r = 0; r < 8; ++r) {
            const int m  = hh * 8 + r;
            const int gm = row0 + m;
            float raw = acc[tt][r];
            float ln  = (raw - red[32 + m]) * red[48 + m] * g + bb;
            float Iin = tanhf(ln) * mag;

            // deterministic noise: popcount of hashed bytes ~ N(0, 0.3)
            unsigned idx = (unsigned)(gm * OUT_F + n);
            unsigned h1 = hashu(idx * 2u + 0x9E3779B9u);
            unsigned h2 = hashu(idx * 2u + 0x85EBCA6Bu);

            float v = -65.0f, u = 0.2f * -65.0f, spikes = 0.0f;
#pragma unroll
            for (int t = 0; t < 8; ++t) {
                unsigned hb = ((t < 4) ? h1 : h2) >> ((t & 3) * 8);
                // popc(8 random bits): mean 4, std sqrt(2); 0.3/sqrt(2)
                float eps = ((float)__popc(hb & 0xFFu) - 4.0f) * 0.212132f;
                float It = Iin + eps;
                float dv = 0.04f * v * v + 5.0f * v + 140.0f - u + It;
                float du = 0.02f * (0.2f * v - u);
                float vn = fminf(fmaxf(v + dv, -100.0f), 60.0f);
                float un = fminf(fmaxf(u + du, -100.0f), 100.0f);
                float spk = (vn >= 30.0f) ? 1.0f : 0.0f;
                v = (vn >= 30.0f) ? -65.0f : vn;
                u = un + 8.0f * spk;
                spikes += spk;
            }
            out[(size_t)gm * OUT_F + n] = spikes * 0.125f;
        }
    }
}

// ---------------------------------------------------------------------------
extern "C" void kernel_launch(void* const* d_in, const int* in_sizes, int n_in,
                              void* d_out, int out_size, void* d_ws, size_t ws_size,
                              hipStream_t stream) {
    (void)in_sizes; (void)n_in; (void)out_size; (void)ws_size;
    const float* x      = (const float*)d_in[0];
    const float* W      = (const float*)d_in[1];
    const float* b_lin  = (const float*)d_in[2];
    const float* gamma  = (const float*)d_in[3];
    const float* beta   = (const float*)d_in[4];
    const float* cmag   = (const float*)d_in[5];
    float* out          = (float*)d_out;

    // d_ws[0 .. 4MB): W as bf16
    unsigned short* Wb = (unsigned short*)d_ws;

    // 2048*1024 elements / (256 threads * 8 per thread) = 1024 blocks
    wconv_kernel<<<(OUT_F * IN_F) / (256 * 8), 256, 0, stream>>>(W, Wb);

    // 4096/16 = 256 blocks of 256 threads (8 waves)
    snn_fused_kernel<<<B_ROWS / BM, NWAVES * 32, 0, stream>>>(
        x, Wb, b_lin, gamma, beta, cmag, out);
}
